// Mamba_layer_20753281974544
// MI455X (gfx1250) — matrix operand order repaired and checked by reading, not compiled, not run on hardware
//
#include <hip/hip_runtime.h>

// ---------------------------------------------------------------- constants
#define DM   256
#define DI   256
#define NS   16
#define BB   4
#define LL   4096
#define MROWS (BB*LL)            // 16384
#define NCHUNK 32
#define CLEN  128                // 32*128 = 4096

typedef __attribute__((ext_vector_type(16))) __bf16        v16bf;
typedef __attribute__((ext_vector_type(8)))  float         v8f;
typedef __attribute__((ext_vector_type(8)))  unsigned int  v8u;

__device__ __forceinline__ unsigned short f2bf(float f) {
    unsigned u = __float_as_uint(f);
    u += 0x7FFFu + ((u >> 16) & 1u);          // round-to-nearest-even
    return (unsigned short)(u >> 16);
}
__device__ __forceinline__ float silu_f(float x) { return x / (1.f + __expf(-x)); }
__device__ __forceinline__ float softplus_f(float x) { return (x > 20.f) ? x : log1pf(__expf(x)); }
__device__ __forceinline__ float gelu_f(float x) {
    float x3 = x * x * x;
    return 0.5f * x * (1.f + tanhf(0.7978845608028654f * (x + 0.044715f * x3)));
}

// ---------------------------------------------------------------- weight prep
// bf16 conversions + fold W_dt into W_x's delta columns:
//   Wbig[:,0:256]   = W_x[:,0:16] @ W_dt        (delta pre-act)
//   Wbig[:,256:288] = W_x[:,16:48]              (B then C)
//   Wbig[:,288:320] = 0                         (N padding for 64-wide tiles)
__global__ __launch_bounds__(256) void prep_weights(
    const float* __restrict__ W_in, const float* __restrict__ W_out,
    const float* __restrict__ W_mlp, const float* __restrict__ W_x,
    const float* __restrict__ W_dt,
    unsigned short* __restrict__ win_bf, unsigned short* __restrict__ wout_bf,
    unsigned short* __restrict__ wmlp_bf, unsigned short* __restrict__ wbig_bf)
{
    int t = blockIdx.x * 256 + threadIdx.x;
    if (t < 131072) { win_bf[t] = f2bf(W_in[t]); return; }
    t -= 131072;
    if (t < 65536) { wout_bf[t] = f2bf(W_out[t]); return; }
    t -= 65536;
    if (t < 65536) { wmlp_bf[t] = f2bf(W_mlp[t]); return; }
    t -= 65536;
    if (t < 256 * 320) {
        int k = t / 320, n = t - k * 320;
        float v;
        if (n < 256) {
            float s = 0.f;
            #pragma unroll
            for (int r = 0; r < 16; ++r) s += W_x[k * 48 + r] * W_dt[r * 256 + n];
            v = s;
        } else if (n < 288) {
            v = W_x[k * 48 + 16 + (n - 256)];
        } else v = 0.f;
        wbig_bf[t] = f2bf(v);
    }
}

// ---------------------------------------------------------------- rmsnorm -> bf16
__global__ __launch_bounds__(256) void rmsnorm_bf16(
    const float* __restrict__ x, const float* __restrict__ w,
    unsigned short* __restrict__ out)
{
    int row = blockIdx.x * 8 + (threadIdx.x >> 5);
    int ln  = threadIdx.x & 31;
    const float* xr = x + (size_t)row * DM + ln * 8;
    float4 a = *(const float4*)(xr);
    float4 b = *(const float4*)(xr + 4);
    float ss = a.x*a.x + a.y*a.y + a.z*a.z + a.w*a.w
             + b.x*b.x + b.y*b.y + b.z*b.z + b.w*b.w;
    #pragma unroll
    for (int off = 16; off; off >>= 1) ss += __shfl_xor(ss, off, 32);
    float sc = rsqrtf(ss * (1.f / DM) + 1e-5f);
    const float* wr = w + ln * 8;
    unsigned short o[8];
    o[0]=f2bf(a.x*sc*wr[0]); o[1]=f2bf(a.y*sc*wr[1]); o[2]=f2bf(a.z*sc*wr[2]); o[3]=f2bf(a.w*sc*wr[3]);
    o[4]=f2bf(b.x*sc*wr[4]); o[5]=f2bf(b.y*sc*wr[5]); o[6]=f2bf(b.z*sc*wr[6]); o[7]=f2bf(b.w*sc*wr[7]);
    *(uint4*)(out + (size_t)row * DM + ln * 8) = *(uint4*)o;
}

// ---------------------------------------------------------------- WMMA GEMM
// C[M,N] = A[M,256](bf16) @ W[256,N](bf16), 128x64 tile, 8 waves/block.
// A tile is staged with CDNA5 async global->LDS DMA (ASYNCcnt path);
// B tile is transposed on the fly through VGPRs into the B-fragment layout.
// MODE 0: out0=xs(f32), out1=silu(res)      (N=512)
// MODE 1: out0=softplus(+b_dt)=delta, out1=B|C (N=288 in 320 alloc)
// MODE 2: out0 = acc + bias + add  (= x1)    (N=256)
// MODE 3: out0 = gelu(acc+bias) + add        (N=256, final output)
#define TLM 128
#define TLN 64
#define LDA 40
#define LDB 34
template<int MODE>
__global__ __launch_bounds__(256) void gemm_bf16(
    const unsigned short* __restrict__ A, const unsigned short* __restrict__ W,
    int Nalloc,
    float* __restrict__ out0, float* __restrict__ out1,
    const float* __restrict__ bias, const float* __restrict__ add)
{
    __shared__ __align__(16) unsigned short sA[TLM * LDA];
    __shared__ __align__(16) unsigned short sB[TLN * LDB];     // transposed: [n][k]
    const int tid = threadIdx.x;
    const int wv  = tid >> 5;
    const int ln  = tid & 31;
    const int nlo = ln & 15;
    const int hf  = ln >> 4;
    const int m0  = blockIdx.x * TLM;
    const int n0  = blockIdx.y * TLN;

    v8f acc[4] = {};

    const int ar = tid >> 1;              // A loader: row 0..127
    const int ac = (tid & 1) * 16;        // 16 bf16 halves
    const int bk = tid >> 3;              // B loader: k row 0..31
    const int bn = (tid & 7) * 8;         // 8 bf16 cols

    // per-lane LDS byte offset for the async A-tile DMA (constant across K)
    const unsigned laoff = (unsigned)(unsigned long long)(void*)&sA[ar * LDA + ac];

    for (int k0 = 0; k0 < 256; k0 += 32) {
        // ---- stage A tile (128x32 bf16): async global->LDS, no VGPR round-trip.
        // INST_OFFSET applies to both LDS and memory sides (ISA 08 §4.4), so the
        // second 16B chunk reuses the same address operands with offset:16.
        {
            unsigned long long ga =
                (unsigned long long)(A + (size_t)(m0 + ar) * 256 + k0 + ac);
            asm volatile(
                "global_load_async_to_lds_b128 %0, %1, off\n\t"
                "global_load_async_to_lds_b128 %0, %1, off offset:16"
                :: "v"(laoff), "v"(ga) : "memory");
        }
        // ---- stage B tile transposed (k fastest) through VGPRs
        uint4 wv4 = *(const uint4*)(W + (size_t)(k0 + bk) * Nalloc + n0 + bn);
        const unsigned short* wss = (const unsigned short*)&wv4;
        #pragma unroll
        for (int e = 0; e < 8; ++e) sB[(bn + e) * LDB + bk] = wss[e];
        if (k0 + 32 < 256) {
            __builtin_prefetch(A + (size_t)(m0 + ar) * 256 + k0 + 32 + ac, 0, 1);
            __builtin_prefetch(W + (size_t)(k0 + 32 + bk) * Nalloc + n0 + bn, 0, 1);
        }
        asm volatile("s_wait_asynccnt 0x0" ::: "memory");   // own async DMAs done
        __syncthreads();                                     // all waves' LDS visible

        // A fragment: lanes 0-15 K{0..15}, lanes 16-31 K{16..31}
        const unsigned short* arow = &sA[((wv << 4) + nlo) * LDA];
        v8u au;
        #pragma unroll
        for (int v = 0; v < 4; ++v) au[v]     = *(const unsigned int*)&arow[16 * hf + 2 * v];
        #pragma unroll
        for (int v = 0; v < 4; ++v) au[4 + v] = *(const unsigned int*)&arow[16 * hf + 8 + 2 * v];
        v16bf av = __builtin_bit_cast(v16bf, au);

        #pragma unroll
        for (int j = 0; j < 4; ++j) {
            // B fragment: VGPR v holds K=2v,2v+1 (lanes 0-15) / K=16+2v.. (16-31)
            const unsigned short* brow = &sB[(16 * j + nlo) * LDB + 16 * hf];
            v8u bu;
            #pragma unroll
            for (int v = 0; v < 8; ++v) bu[v] = *(const unsigned int*)&brow[2 * v];
            v16bf bv = __builtin_bit_cast(v16bf, bu);
            acc[j] = __builtin_amdgcn_wmma_f32_16x16x32_bf16(
                         false, av, false, bv, (short)0, acc[j], false, false);
        }
        __syncthreads();
    }

    // epilogue: D element (m = m0 + 16*wv + 8*hf + g, n = n0 + 16*j + nlo)
    #pragma unroll
    for (int j = 0; j < 4; ++j) {
        const int n = n0 + 16 * j + nlo;
        #pragma unroll
        for (int g = 0; g < 8; ++g) {
            const int m = m0 + (wv << 4) + (hf << 3) + g;
            float v = acc[j][g];
            if constexpr (MODE == 0) {
                if (n < 256) out0[(size_t)m * DI + n] = v;
                else         out1[(size_t)m * DI + (n - 256)] = silu_f(v);
            } else if constexpr (MODE == 1) {
                if (n < 256)      out0[(size_t)m * DI + n] = softplus_f(v + bias[n]);
                else if (n < 288) out1[(size_t)m * 32 + (n - 256)] = v;
            } else if constexpr (MODE == 2) {
                out0[(size_t)m * DM + n] = v + bias[n] + add[(size_t)m * DM + n];
            } else {
                out0[(size_t)m * DM + n] = gelu_f(v + bias[n]) + add[(size_t)m * DM + n];
            }
        }
    }
}

// ---------------------------------------------------------------- depthwise conv + SiLU
__global__ __launch_bounds__(256) void conv_silu(
    const float* __restrict__ xs, const float* __restrict__ cw,
    const float* __restrict__ cb, unsigned short* __restrict__ xact_bf,
    float* __restrict__ u)
{
    int idx = blockIdx.x * 256 + threadIdx.x;       // (b,l,d) flat, d fastest
    int d  = idx & (DI - 1);
    int l  = (idx >> 8) & (LL - 1);
    const float* base = xs + (size_t)idx;
    float acc = cb[d];
    float w0 = cw[d*4+0], w1 = cw[d*4+1], w2 = cw[d*4+2], w3 = cw[d*4+3];
    if (l >= 3) acc += w0 * base[-3 * DI];
    if (l >= 2) acc += w1 * base[-2 * DI];
    if (l >= 1) acc += w2 * base[-1 * DI];
    acc += w3 * base[0];
    float a = silu_f(acc);
    u[idx] = a;
    xact_bf[idx] = f2bf(a);
}

// ---------------------------------------------------------------- chunked selective scan
// wave covers (b, d-pair, chunk): lane = n + 16*(d&1)
__device__ __forceinline__ void scan_decode(int& b, int& d, int& n, int& c) {
    int Wg = blockIdx.x * 8 + (threadIdx.x >> 5);
    int ln = threadIdx.x & 31;
    n = ln & 15;
    c = Wg & (NCHUNK - 1);
    int dp = (Wg >> 5) & 127;
    b = Wg >> 12;
    d = dp * 2 + (ln >> 4);
}

__global__ __launch_bounds__(256) void scan_pass1(
    const float* __restrict__ delta, const float* __restrict__ u,
    const float* __restrict__ bc, const float* __restrict__ A_log,
    float* __restrict__ h_end, float* __restrict__ sumdA)
{
    int b, d, n, c; scan_decode(b, d, n, c);
    float Ad = -__expf(A_log[d * NS + n]);
    float h = 0.f, sa = 0.f;
    size_t base = (size_t)b * LL + (size_t)c * CLEN;
    for (int i = 0; i < CLEN; ++i) {
        size_t bl = base + i;
        float dt = delta[bl * DI + d];
        float dA = dt * Ad;
        sa += dA;
        h = __expf(dA) * h + dt * u[bl * DI + d] * bc[bl * 32 + n];
    }
    size_t idx = (((size_t)b * DI + d) * NS + n) * NCHUNK + c;
    h_end[idx] = h;
    sumdA[idx] = sa;
}

__global__ __launch_bounds__(256) void scan_pass2(
    const float* __restrict__ h_end, const float* __restrict__ sumdA,
    float* __restrict__ h_init)
{
    int t = blockIdx.x * 256 + threadIdx.x;         // B*DI*NS = 16384
    size_t base = (size_t)t * NCHUNK;
    float h = 0.f;
    for (int c = 0; c < NCHUNK; ++c) {
        h_init[base + c] = h;
        h = __expf(sumdA[base + c]) * h + h_end[base + c];
    }
}

__global__ __launch_bounds__(256) void scan_pass3(
    const float* __restrict__ delta, const float* __restrict__ u,
    const float* __restrict__ bc, const float* __restrict__ A_log,
    const float* __restrict__ h_init, const float* __restrict__ Dv,
    const float* __restrict__ sres, unsigned short* __restrict__ yg)
{
    int b, d, n, c; scan_decode(b, d, n, c);
    float Ad = -__expf(A_log[d * NS + n]);
    size_t idx = (((size_t)b * DI + d) * NS + n) * NCHUNK + c;
    float h  = h_init[idx];
    float Dd = Dv[d];
    size_t base = (size_t)b * LL + (size_t)c * CLEN;
    for (int i = 0; i < CLEN; ++i) {
        size_t bl = base + i;
        float dt = delta[bl * DI + d];
        float uu = u[bl * DI + d];
        h = __expf(dt * Ad) * h + dt * uu * bc[bl * 32 + n];
        float y = bc[bl * 32 + 16 + n] * h;         // C * h
        y += __shfl_xor(y, 1, 32);
        y += __shfl_xor(y, 2, 32);
        y += __shfl_xor(y, 4, 32);
        y += __shfl_xor(y, 8, 32);                  // stays within 16-lane group
        if (n == 0) {
            float g = (y + uu * Dd) * sres[bl * DI + d];
            yg[bl * DI + d] = f2bf(g);
        }
    }
}

// ---------------------------------------------------------------- launcher
extern "C" void kernel_launch(void* const* d_in, const int* in_sizes, int n_in,
                              void* d_out, int out_size, void* d_ws, size_t ws_size,
                              hipStream_t stream) {
    const float* inputs  = (const float*)d_in[0];
    const float* rms1_w  = (const float*)d_in[1];
    const float* rms2_w  = (const float*)d_in[2];
    const float* W_in    = (const float*)d_in[3];
    const float* conv_w  = (const float*)d_in[4];
    const float* conv_b  = (const float*)d_in[5];
    const float* W_x     = (const float*)d_in[6];
    const float* W_dt    = (const float*)d_in[7];
    const float* b_dt    = (const float*)d_in[8];
    const float* A_log   = (const float*)d_in[9];
    const float* D_param = (const float*)d_in[10];
    const float* W_out   = (const float*)d_in[11];
    const float* b_out   = (const float*)d_in[12];
    const float* W_mlp   = (const float*)d_in[13];
    const float* b_mlp   = (const float*)d_in[14];

    char* ws = (char*)d_ws;
    const size_t SZ_M256F = (size_t)MROWS * 256 * 4;   // 16 MB
    const size_t SZ_M256H = (size_t)MROWS * 256 * 2;   //  8 MB
    size_t off = 0;
    unsigned short* win_bf  = (unsigned short*)(ws + off); off += 256 * 512 * 2;
    unsigned short* wout_bf = (unsigned short*)(ws + off); off += 256 * 256 * 2;
    unsigned short* wmlp_bf = (unsigned short*)(ws + off); off += 256 * 256 * 2;
    unsigned short* wbig_bf = (unsigned short*)(ws + off); off += 256 * 320 * 2;
    off = (off + 255) & ~(size_t)255;
    unsigned short* xnorm_bf = (unsigned short*)(ws + off); off += SZ_M256H; // reused as ygated
    float* xs_f32   = (float*)(ws + off); off += SZ_M256F;                   // reused as delta
    float* sres_f32 = (float*)(ws + off); off += SZ_M256F;
    unsigned short* xact_bf = (unsigned short*)(ws + off); off += SZ_M256H;  // reused as x1norm
    float* u_f32    = (float*)(ws + off); off += SZ_M256F;
    float* bc_f32   = (float*)(ws + off); off += (size_t)MROWS * 32 * 4;
    float* h_end    = (float*)(ws + off); off += (size_t)BB * DI * NS * NCHUNK * 4;
    float* sumdA    = (float*)(ws + off); off += (size_t)BB * DI * NS * NCHUNK * 4;
    float* h_init   = (float*)(ws + off); off += (size_t)BB * DI * NS * NCHUNK * 4;
    float* x1_f32   = (float*)(ws + off); off += SZ_M256F;
    float* delta_f32 = xs_f32;            // safe reuse (xs consumed by conv)
    unsigned short* yg_bf  = xnorm_bf;    // safe reuse (xnorm consumed by GEMM1)
    unsigned short* x1n_bf = xact_bf;     // safe reuse (xact consumed by GEMM_x)
    (void)off; (void)ws_size; (void)in_sizes; (void)n_in; (void)out_size;

    prep_weights<<<1344, 256, 0, stream>>>(W_in, W_out, W_mlp, W_x, W_dt,
                                           win_bf, wout_bf, wmlp_bf, wbig_bf);
    rmsnorm_bf16<<<MROWS / 8, 256, 0, stream>>>(inputs, rms1_w, xnorm_bf);
    gemm_bf16<0><<<dim3(MROWS / TLM, 512 / TLN), 256, 0, stream>>>(
        xnorm_bf, win_bf, 512, xs_f32, sres_f32, nullptr, nullptr);
    conv_silu<<<(MROWS * DI) / 256, 256, 0, stream>>>(xs_f32, conv_w, conv_b, xact_bf, u_f32);
    gemm_bf16<1><<<dim3(MROWS / TLM, 320 / TLN), 256, 0, stream>>>(
        xact_bf, wbig_bf, 320, delta_f32, bc_f32, b_dt, nullptr);
    scan_pass1<<<(BB * (DI / 2) * NCHUNK) / 8, 256, 0, stream>>>(
        delta_f32, u_f32, bc_f32, A_log, h_end, sumdA);
    scan_pass2<<<(BB * DI * NS) / 256, 256, 0, stream>>>(h_end, sumdA, h_init);
    scan_pass3<<<(BB * (DI / 2) * NCHUNK) / 8, 256, 0, stream>>>(
        delta_f32, u_f32, bc_f32, A_log, h_init, D_param, sres_f32, yg_bf);
    gemm_bf16<2><<<dim3(MROWS / TLM, 256 / TLN), 256, 0, stream>>>(
        yg_bf, wout_bf, 256, x1_f32, nullptr, b_out, inputs);
    rmsnorm_bf16<<<MROWS / 8, 256, 0, stream>>>(x1_f32, rms2_w, x1n_bf);
    gemm_bf16<3><<<dim3(MROWS / TLM, 256 / TLN), 256, 0, stream>>>(
        x1n_bf, wmlp_bf, 256, (float*)d_out, nullptr, b_mlp, x1_f32);
}
